// HoRALinear_84439057039540
// MI455X (gfx1250) — matrix operand order repaired and checked by reading, not compiled, not run on hardware
//
#include <hip/hip_runtime.h>
#include <hip/hip_bf16.h>
#include <math.h>
#include <stdint.h>

typedef _Float16 v16h __attribute__((ext_vector_type(16)));
typedef _Float16 v8h  __attribute__((ext_vector_type(8)));
typedef _Float16 v4h  __attribute__((ext_vector_type(4)));
typedef float    v8f  __attribute__((ext_vector_type(8)));

#define OUTF  4096
#define INF   4096
#define RNK   16
#define NFREQ 10000
#define MTOT  8192          // 4 * 2048
#define PI2   6.28318530717958647692f

// ---------------------------------------------------------------------------
// Stage 0: dense_spectrum[4096][16]; flat index == idx (row*16+col == idx)
// ---------------------------------------------------------------------------
__global__ void zero_f32_kernel(float* __restrict__ p, int n) {
    int i = blockIdx.x * blockDim.x + threadIdx.x;
    if (i < n) p[i] = 0.0f;
}

__global__ void scatter_kernel(const float* __restrict__ spec,
                               const int* __restrict__ idx,
                               float* __restrict__ dense, int n) {
    int i = blockIdx.x * blockDim.x + threadIdx.x;
    if (i < n) dense[idx[i]] = spec[i];
}

// ---------------------------------------------------------------------------
// Stage 1: 1024-point IDFT along axis -2 of [4][1024][16] chunks.
// T[d][m][l] = (1/1024) * sum_k S[d][k][l] * exp(+2*pi*i*m*k/1024)
// LDS twiddle table; phase tracked by integer add (no trig in the loop).
// ---------------------------------------------------------------------------
__global__ void idft_axis0_kernel(const float* __restrict__ dense,
                                  float2* __restrict__ T) {
    __shared__ float2 tw[1024];
    const int tid = threadIdx.x;
    for (int j = tid; j < 1024; j += 256) {
        float s, c;
        __sincosf(PI2 * (float)j * (1.0f / 1024.0f), &s, &c);
        tw[j] = make_float2(c, s);
    }
    __syncthreads();

    const int d = blockIdx.x >> 6;
    const int m = ((blockIdx.x & 63) << 4) + (tid >> 4);
    const int l = tid & 15;
    const float* S = dense + d * 16384 + l;

    float re = 0.0f, im = 0.0f;
    int ph = 0;
    for (int k = 0; k < 1024; ++k) {
        const float s = S[k * 16];
        if (s != 0.0f) {
            const float2 w = tw[ph];
            re = fmaf(s, w.x, re);
            im = fmaf(s, w.y, im);
        }
        ph = (ph + m) & 1023;
    }
    T[(d * 1024 + m) * 16 + l] =
        make_float2(re * (1.0f / 1024.0f), im * (1.0f / 1024.0f));
}

// ---------------------------------------------------------------------------
// Stage 2: 16-point IDFT along last axis, keep real part.
// ---------------------------------------------------------------------------
__global__ void idft_axis1_kernel(const float2* __restrict__ T,
                                  float* __restrict__ Bm) {
    const int g = blockIdx.x * 256 + threadIdx.x;   // 65536 total
    const int o = g >> 4;
    const int n = g & 15;
    const float2* t = T + o * 16;
    float acc = 0.0f;
#pragma unroll
    for (int l = 0; l < 16; ++l) {
        float s, c;
        __sincosf(PI2 * (float)(n * l) * (1.0f / 16.0f), &s, &c);
        acc += t[l].x * c - t[l].y * s;
    }
    Bm[g] = acc * (1.0f / 16.0f);
}

// ---------------------------------------------------------------------------
// Stage 3: fold weights.  Wh[o][i] = f16( base_w[o][i] + 150*sum_r B[o][r]*A[r][i] )
// ---------------------------------------------------------------------------
__global__ void fold_weight_kernel(const float* __restrict__ base_w,
                                   const float* __restrict__ Bm,
                                   const float* __restrict__ hora_A,
                                   _Float16* __restrict__ Wh) {
    const int o = blockIdx.x;
    __shared__ float br[16];
    if (threadIdx.x < 16) br[threadIdx.x] = Bm[o * 16 + threadIdx.x] * 150.0f;
    __syncthreads();
    for (int i = threadIdx.x; i < INF; i += 256) {
        float acc = base_w[(size_t)o * INF + i];
#pragma unroll
        for (int r = 0; r < RNK; ++r)
            acc = fmaf(br[r], hora_A[r * INF + i], acc);
        Wh[(size_t)o * INF + i] = (_Float16)acc;
    }
}

// ---------------------------------------------------------------------------
// Stage 4: main GEMM.  out[m][o] = sum_k x[m][k] * Wh[o][k] + bias[o]
// 128x128 block tile, BK=32, double-buffered LDS.
// W tile: GLOBAL_LOAD_ASYNC_TO_LDS_B128 (ASYNCcnt), memory -> LDS direct.
// x tile: global f32 loads issued pre-compute, cvt->f16 ds_store post-compute.
// 8 wave32 waves, each wave = 64(M) x 32(N) = 4x2 WMMA f32_16x16x32_f16.
// ---------------------------------------------------------------------------
#define BM 128
#define BN 128
#define BK 32
#define KP (BK + 8)          // 80B row stride (16B aligned, bank-skewed)
#define NT (INF / BK)        // 128 k-tiles

__global__ __launch_bounds__(256)
void hora_gemm_kernel(const float* __restrict__ xg,
                      const _Float16* __restrict__ Wh,
                      const float* __restrict__ bias,
                      float* __restrict__ out) {
    __shared__ _Float16 sA[2][BM][KP];   // x tile (M x K) f16, double buffered
    __shared__ _Float16 sB[2][BN][KP];   // W tile (N x K) f16, double buffered

    const int tid  = threadIdx.x;
    const int lane = tid & 31;
    const int w    = tid >> 5;
    const int wm   = (w & 1) * 64;
    const int wn   = (w >> 1) * 32;
    const int m0   = blockIdx.y * BM;
    const int o0   = blockIdx.x * BN;

    // A fragment: lane<16 -> K {0..7,16..23}; lane>=16 -> K {8..15,24..31}
    const int ka = (lane >> 4) * 8;
    // B fragment: lane<16 -> K {0..7,16..23}; lane>=16 -> K {8..15,24..31} (N = lane&15)
    const int ko = (lane >> 4) * 8;
    const int ml = lane & 15;

    const uint64_t wh_base = (uint64_t)(uintptr_t)Wh;
    const uint32_t ldsB[2] = { (uint32_t)(uintptr_t)&sB[0][0][0],
                               (uint32_t)(uintptr_t)&sB[1][0][0] };

    // ---- async W tile: 128x32 f16 = 512 x b128; 2 issues/thread
    auto issue_w = [&](int p, int kt) {
#pragma unroll
        for (int i = 0; i < 2; ++i) {
            const int li = tid + i * 256;
            const int r  = li >> 2;        // 4 b128 per row
            const int c8 = li & 3;
            const uint32_t goff  = (uint32_t)((((o0 + r) << 12) + kt + c8 * 8) * 2);
            const uint32_t laddr = ldsB[p] + (uint32_t)((r * KP + c8 * 8) * 2);
            asm volatile("global_load_async_to_lds_b128 %0, %1, %2"
                         :: "v"(laddr), "v"(goff), "s"(wh_base) : "memory");
        }
    };
    // ---- x tile: 128x32 f32 = 1024 x float4; 4 loads/thread
    auto load_x = [&](int kt, float4* xr) {
#pragma unroll
        for (int i = 0; i < 4; ++i) {
            const int li = tid + i * 256;
            const int r  = li >> 3;        // 8 float4 per row
            const int c4 = li & 7;
            xr[i] = *(const float4*)&xg[(size_t)(m0 + r) * INF + kt + c4 * 4];
        }
    };
    auto store_x = [&](int p, const float4* xr) {
#pragma unroll
        for (int i = 0; i < 4; ++i) {
            const int li = tid + i * 256;
            const int r  = li >> 3;
            const int c4 = li & 7;
            v4h h;
            h[0] = (_Float16)xr[i].x; h[1] = (_Float16)xr[i].y;
            h[2] = (_Float16)xr[i].z; h[3] = (_Float16)xr[i].w;
            *(v4h*)&sA[p][r][c4 * 4] = h;
        }
    };

    const v8f vzero = {0.f, 0.f, 0.f, 0.f, 0.f, 0.f, 0.f, 0.f};
    v8f acc[4][2];
#pragma unroll
    for (int i = 0; i < 4; ++i)
#pragma unroll
        for (int j = 0; j < 2; ++j) acc[i][j] = vzero;

    // ---- prologue: stage tile 0 into buffer 0
    issue_w(0, 0);
    {
        float4 xr[4];
        load_x(0, xr);
        store_x(0, xr);
    }
    asm volatile("s_wait_asynccnt 0x0" ::: "memory");
    __syncthreads();

    // ---- pipelined main loop
    for (int t = 0; t < NT; ++t) {
        const int p = t & 1;
        const bool more = (t + 1 < NT);
        float4 xr[4];
        if (more) {
            issue_w(1 - p, (t + 1) * BK);      // async copy next W tile
            load_x((t + 1) * BK, xr);          // issue next x loads (regs)
        }

        // compute on buffer p (one kk step: BK == 32)
        v16h af[4];
#pragma unroll
        for (int mf = 0; mf < 4; ++mf) {
            const int row = wm + mf * 16 + ml;
            const v8h lo = *(const v8h*)&sA[p][row][ka];
            const v8h hi = *(const v8h*)&sA[p][row][ka + 16];
#pragma unroll
            for (int e = 0; e < 8; ++e) { af[mf][e] = lo[e]; af[mf][e + 8] = hi[e]; }
        }
        v16h bf[2];
#pragma unroll
        for (int nf = 0; nf < 2; ++nf) {
            const int row = wn + nf * 16 + ml;
            const v8h lo = *(const v8h*)&sB[p][row][ko];
            const v8h hi = *(const v8h*)&sB[p][row][ko + 16];
#pragma unroll
            for (int e = 0; e < 8; ++e) { bf[nf][e] = lo[e]; bf[nf][e + 8] = hi[e]; }
        }
#pragma unroll
        for (int mf = 0; mf < 4; ++mf)
#pragma unroll
            for (int nf = 0; nf < 2; ++nf)
                acc[mf][nf] = __builtin_amdgcn_wmma_f32_16x16x32_f16(
                    false, af[mf], false, bf[nf],
                    (short)0, acc[mf][nf], false, false);

        if (more) store_x(1 - p, xr);          // cvt + ds_store next x tile
        asm volatile("s_wait_asynccnt 0x0" ::: "memory");
        __syncthreads();
    }

    // ---- epilogue: C/D element r -> (M = r + 8*(lane>=16), N = lane&15)
    const int cm = (lane >> 4) * 8;
    const int cn = lane & 15;
#pragma unroll
    for (int nf = 0; nf < 2; ++nf) {
        const int oc = o0 + wn + nf * 16 + cn;
        const float bv = bias[oc];
#pragma unroll
        for (int mf = 0; mf < 4; ++mf) {
#pragma unroll
            for (int r = 0; r < 8; ++r) {
                out[(size_t)(m0 + wm + mf * 16 + cm + r) * OUTF + oc] =
                    acc[mf][nf][r] + bv;
            }
        }
    }
}

// ---------------------------------------------------------------------------
// Host launcher
// ---------------------------------------------------------------------------
extern "C" void kernel_launch(void* const* d_in, const int* in_sizes, int n_in,
                              void* d_out, int out_size, void* d_ws, size_t ws_size,
                              hipStream_t stream) {
    const float* x      = (const float*)d_in[0];   // [4,2048,4096]
    const float* base_w = (const float*)d_in[1];   // [4096,4096]
    const float* base_b = (const float*)d_in[2];   // [4096]
    const float* spec   = (const float*)d_in[3];   // [10000]
    const float* hora_A = (const float*)d_in[4];   // [16,4096]
    const int*   idx    = (const int*)d_in[5];     // [10000]
    float* out = (float*)d_out;

    char* ws = (char*)d_ws;
    float*    dense = (float*)ws;                                   // 256 KB
    float2*   T     = (float2*)(ws + 65536u * 4);                   // 512 KB
    float*    Bm    = (float*)(ws + 65536u * 4 + 65536u * 8);       // 256 KB
    _Float16* Wh    = (_Float16*)(ws + 65536u * 4 + 65536u * 8 + 65536u * 4); // 32 MB

    zero_f32_kernel<<<65536 / 256, 256, 0, stream>>>(dense, 65536);
    scatter_kernel<<<(NFREQ + 255) / 256, 256, 0, stream>>>(spec, idx, dense, NFREQ);
    idft_axis0_kernel<<<256, 256, 0, stream>>>(dense, T);
    idft_axis1_kernel<<<256, 256, 0, stream>>>(T, Bm);
    fold_weight_kernel<<<OUTF, 256, 0, stream>>>(base_w, Bm, hora_A, Wh);

    dim3 grid(OUTF / BN, MTOT / BM);
    hora_gemm_kernel<<<grid, 256, 0, stream>>>(x, Wh, base_b, out);
    (void)in_sizes; (void)n_in; (void)out_size; (void)ws_size;
}
